// Model_PDE_2_67362267070602
// MI455X (gfx1250) — matrix operand order repaired and checked by reading, not compiled, not run on hardware
//
#include <hip/hip_runtime.h>

typedef __attribute__((ext_vector_type(16))) _Float16 v16h;
typedef __attribute__((ext_vector_type(8)))  float    v8f;

#define N_SAMPLES 131072
#define HDIM 1024

#if __has_builtin(__builtin_amdgcn_tanhf)
__device__ __forceinline__ float fast_tanh(float v) { return __builtin_amdgcn_tanhf(v); }
#else
__device__ __forceinline__ float fast_tanh(float v) { return tanhf(v); }
#endif

// One wave handles a 16-sample tile; 8 waves (256 threads) per block -> 128 samples/block.
__global__ __launch_bounds__(256) void pde_wmma_kernel(
    const float* __restrict__ x,  const float* __restrict__ W1,
    const float* __restrict__ b1, const float* __restrict__ w2,
    const float* __restrict__ b2, float* __restrict__ out)
{
    // Staged weights (f32 for the tanh argument, f16 tables for the WMMA B operands)
    __shared__ float sW1[HDIM * 2];   // interleaved W1[h][0], W1[h][1]
    __shared__ float sB1[HDIM];
    __shared__ __attribute__((aligned(32))) _Float16 tW2[HDIM];  // w2
    __shared__ __attribute__((aligned(32))) _Float16 tG0[HDIM];  // w2*W1_0
    __shared__ __attribute__((aligned(32))) _Float16 tG1[HDIM];  // w2*W1_1
    __shared__ __attribute__((aligned(32))) _Float16 tUU[HDIM];  // -2*w2*W1_1^2
    __shared__ __attribute__((aligned(32))) _Float16 tZZ[HDIM];  // zeros (inactive B columns)
    __shared__ float red[8][16][4];   // per-wave r0..r3 per sample row

    const int tid = threadIdx.x;
    for (int h = tid; h < HDIM; h += 256) {
        float a = W1[2 * h], bb = W1[2 * h + 1], w = w2[h];
        sW1[2 * h] = a; sW1[2 * h + 1] = bb; sB1[h] = b1[h];
        tW2[h] = (_Float16)w;
        tG0[h] = (_Float16)(w * a);
        tG1[h] = (_Float16)(w * bb);
        tUU[h] = (_Float16)(-2.0f * w * bb * bb);
        tZZ[h] = (_Float16)0.0f;
    }
    __syncthreads();

    const int wave    = tid >> 5;
    const int lane    = tid & 31;
    const int m       = lane & 15;   // sample-row within tile / B column
    const int halfsel = lane >> 4;
    const int sample  = (blockIdx.x * 8 + wave) * 16 + m;

    const float x0 = x[2 * sample];
    const float x1 = x[2 * sample + 1];

    // Loop-invariant B base pointers: real table for this lane's active column,
    // zero table otherwise. Keeps the inner loop branch-free with EXEC == all-1s.
    const _Float16* bpt = (m == 0) ? tW2 : tZZ;
    const _Float16* bps = (m == 1) ? tG0 : (m == 2) ? tG1 : tZZ;
    const _Float16* bpu = (m == 3) ? tUU : tZZ;

    v8f acc = {};

    for (int kc = 0; kc < HDIM; kc += 32) {
        // ---- A operands, contiguous K run = kc + 16*halfsel .. +15 (matches B) ----
        v16h At, As, Au;
#pragma unroll
        for (int j = 0; j < 8; ++j) {
            const int k0 = kc + halfsel * 16 + 2 * j;
#pragma unroll
            for (int p = 0; p < 2; ++p) {
                const int k = k0 + p;
                float z  = fmaf(x0, sW1[2 * k], fmaf(x1, sW1[2 * k + 1], sB1[k]));
                float t  = fast_tanh(z);
                float om = fmaf(-t, t, 1.0f);       // 1 - t^2
                At[2 * j + p] = (_Float16)t;
                As[2 * j + p] = (_Float16)om;
                Au[2 * j + p] = (_Float16)(t * om);
            }
        }
        // ---- B operands: column m, contiguous K run = kc + 16*halfsel .. +15 ----
        const int boff = kc + halfsel * 16;
        v16h Bt = *(const v16h*)&bpt[boff];
        v16h Bs = *(const v16h*)&bps[boff];
        v16h Bu = *(const v16h*)&bpu[boff];

        acc = __builtin_amdgcn_wmma_f32_16x16x32_f16(false, At, false, Bt, (short)0, acc, false, false);
        acc = __builtin_amdgcn_wmma_f32_16x16x32_f16(false, As, false, Bs, (short)0, acc, false, false);
        acc = __builtin_amdgcn_wmma_f32_16x16x32_f16(false, Au, false, Bu, (short)0, acc, false, false);
    }

    // D layout: lane n (0-15) holds column n, rows 0-7 in acc[0..7]; lanes 16-31 rows 8-15.
    if (m < 4) {
#pragma unroll
        for (int r = 0; r < 8; ++r)
            red[wave][r + 8 * halfsel][m] = acc[r];
    }
    __syncthreads();

    if (lane < 16) {
        const int s_idx = (blockIdx.x * 8 + wave) * 16 + lane;
        const float r0 = red[wave][lane][0];
        const float r1 = red[wave][lane][1];
        const float r2 = red[wave][lane][2];
        const float r3 = red[wave][lane][3];
        const float xx1 = x[2 * s_idx + 1];

        out[s_idx] = r0 + b2[0];
        // pde = b_f*x1^2 + df_dt + 0.5*sigma^2*df_dxdx + b*x1*df_dx - c^2/(4 c_f)*df_dx^2
        float pde = 0.5f * xx1 * xx1 + r1 + 0.5f * r3
                  + 0.5f * xx1 * r2 - (0.25f / 3.6f) * r2 * r2;
        out[N_SAMPLES + s_idx] = pde;
    }
}

extern "C" void kernel_launch(void* const* d_in, const int* in_sizes, int n_in,
                              void* d_out, int out_size, void* d_ws, size_t ws_size,
                              hipStream_t stream) {
    const float* x  = (const float*)d_in[0];
    const float* W1 = (const float*)d_in[1];
    const float* b1 = (const float*)d_in[2];
    const float* w2 = (const float*)d_in[3];
    const float* b2 = (const float*)d_in[4];
    float* out = (float*)d_out;

    dim3 grid(N_SAMPLES / 128);  // 1024 blocks, 8 waves each, 16 samples per wave
    pde_wmma_kernel<<<grid, 256, 0, stream>>>(x, W1, b1, w2, b2, out);
}
